// MultiHeadAttentionLayer_51290499449233
// MI455X (gfx1250) — compile-verified, matrix operand order checked by reading, not run on hardware
//
#include <hip/hip_runtime.h>
#include <hip/hip_bf16.h>

// ---------------------------------------------------------------------------
// Swin shifted-window attention for MI455X (gfx1250), bf16 WMMA pipeline.
//   K1: qkv = x @ w_qkv + b_qkv   (async-stage x slab -> LDS, bf16 WMMA)
//   K2: per-window attention      (async-stage V, WMMA QK^T + PV, f32 softmax)
//   K3: out = attn @ w_out + b_out (async-stage attn slab -> LDS, bf16 WMMA)
// ---------------------------------------------------------------------------

typedef __attribute__((ext_vector_type(16))) __bf16 v16bf;
typedef __attribute__((ext_vector_type(8)))  float  v8f;
typedef __attribute__((ext_vector_type(2)))  __bf16 bf16x2;
typedef __attribute__((ext_vector_type(2)))  float  f32x2;

#define Bd   8
#define Hd   256
#define Wd   256
#define Dd   96
#define NH   6
#define DK   16
#define SH_  4
#define HW   (Hd * Wd)          // 65536
#define MROWS (Bd * HW)         // 524288

// K index for 16-bit A/B WMMA operand: VGPR-pair p (0..7), half = lane>=16.
__device__ __forceinline__ int kofs(int p, int half) {
    return ((p < 4) ? (2 * p) : (16 + 2 * (p - 4))) + 8 * half;
}

// Async copy of 16 bytes global -> LDS (per-lane addresses), ASYNCcnt-tracked.
__device__ __forceinline__ void async_copy_b128(unsigned lds_off, const void* g) {
    unsigned long long ga = (unsigned long long)(size_t)g;
    asm volatile("global_load_async_to_lds_b128 %0, %1, off"
                 :: "v"(lds_off), "v"(ga) : "memory");
}
__device__ __forceinline__ void async_wait0() {
    asm volatile("s_wait_asynccnt 0" ::: "memory");
}
// Generic shared-pointer -> 32-bit LDS byte offset (flat aperture: addr[31:0]).
__device__ __forceinline__ unsigned lds_offset(const void* p) {
    return (unsigned)(size_t)p;
}

// ---------------------------------------------------------------------------
// Kernel 1: QKV projection.  M=524288, K=96, N=288.
// One block (192 thr, 6 waves) owns a 64-row x-slab (LDS) and all 18 N-tiles:
// wave w -> ntiles {w, w+6, w+12}, each reused across 4 M-strips.
// Output layout: qkv[which(3)][b][head][y][x][dk16]  (bf16)
// ---------------------------------------------------------------------------
#define K1_PAD 100   // f32 per padded LDS row (400B: 16B-aligned, bank-clean)

__global__ __launch_bounds__(192) void qkv_proj_kernel(
    const float* __restrict__ x, const float* __restrict__ w_qkv,
    const float* __restrict__ b_qkv, __bf16* __restrict__ qkv)
{
    __shared__ float sX[64 * K1_PAD];
    const int tid  = threadIdx.x;
    const int lane = tid & 31;
    const int wave = tid >> 5;
    const int half = lane >> 4;
    const int l16  = lane & 15;
    const int row0 = blockIdx.x * 64;

    __builtin_prefetch(x + (size_t)(row0 + 64) * Dd + tid * 32, 0, 0);

    // Async-stage 64 rows x 96 f32: 1536 x 16B chunks over 192 threads.
    #pragma unroll
    for (int i = 0; i < 8; ++i) {
        const int c     = tid + i * 192;
        const int row   = c / 24;
        const int off16 = c % 24;            // 16B chunk within row
        async_copy_b128(lds_offset(&sX[row * K1_PAD + off16 * 4]),
                        x + (size_t)(row0 + row) * Dd + off16 * 4);
    }
    async_wait0();
    __syncthreads();

    const int bb = row0 >> 16;               // batch (64-row slab never crosses)
    #pragma unroll
    for (int t = 0; t < 3; ++t) {
        const int nt   = wave + t * 6;
        const int colB = nt * 16 + l16;

        // B operand (3 K-steps) loaded once, reused for 4 M-strips.
        v16bf bmat[3];
        #pragma unroll
        for (int kk = 0; kk < 3; ++kk)
            #pragma unroll
            for (int p = 0; p < 8; ++p) {
                const int k = kofs(p, half) + 32 * kk;
                bmat[kk][2 * p]     = (__bf16)w_qkv[(size_t)(k)     * 288 + colB];
                bmat[kk][2 * p + 1] = (__bf16)w_qkv[(size_t)(k + 1) * 288 + colB];
            }
        const float bias = b_qkv[colB];
        const int which  = colB / Dd;
        const int head   = (colB % Dd) >> 4;
        const int dk     = colB & 15;
        const size_t dstbase =
            (((size_t)which * Bd + bb) * NH + head) * ((size_t)HW * DK);

        #pragma unroll
        for (int ms = 0; ms < 4; ++ms) {
            const int arow = ms * 16 + l16;
            v8f c = {};
            #pragma unroll
            for (int kk = 0; kk < 3; ++kk) {
                v16bf a;
                #pragma unroll
                for (int p = 0; p < 8; ++p) {
                    const int k = kofs(p, half) + 32 * kk;
                    f32x2 av = *(const f32x2*)(&sX[arow * K1_PAD + k]);
                    a[2 * p]     = (__bf16)av.x;
                    a[2 * p + 1] = (__bf16)av.y;
                }
                c = __builtin_amdgcn_wmma_f32_16x16x32_bf16(
                        false, a, false, bmat[kk], (short)0, c, false, false);
            }
            #pragma unroll
            for (int r = 0; r < 8; ++r) {
                const int row = row0 + ms * 16 + r + 8 * half;
                const int pix = row & (HW - 1);
                qkv[dstbase + (size_t)pix * DK + dk] = (__bf16)(c[r] + bias);
            }
        }
    }
}

// ---------------------------------------------------------------------------
// Kernel 2: one block = one (batch, head, window). 128 threads = 4 waves.
// Each wave owns a 16-row strip of the 64x64 score matrix.
// ---------------------------------------------------------------------------
#define SP_PAD 68   // bf16 per padded P row (34 dwords: bank-clean regather)

__global__ __launch_bounds__(128) void win_attn_kernel(
    const __bf16* __restrict__ qkv, const float* __restrict__ rpe_table,
    const int* __restrict__ rp_index, const unsigned char* __restrict__ mask,
    __bf16* __restrict__ attn_out)
{
    __shared__ __bf16 sV[64 * DK];       // V row-major: [token][dk]
    __shared__ __bf16 sP[64 * SP_PAD];   // probabilities, padded rows

    const int tid  = threadIdx.x;
    const int lane = tid & 31;
    const int wave = tid >> 5;
    const int half = lane >> 4;
    const int l16  = lane & 15;

    const int idx  = blockIdx.x;
    const int wj   = idx & 31;
    const int wi   = (idx >> 5) & 31;
    const int head = (idx >> 10) % NH;
    const int b    = idx / (NH * 32 * 32);

    const size_t plane = (size_t)HW * DK;
    const size_t qbase = (((size_t)0 * Bd + b) * NH + head) * plane;
    const size_t kbase = (((size_t)1 * Bd + b) * NH + head) * plane;
    const size_t vbase = (((size_t)2 * Bd + b) * NH + head) * plane;

    // token (0..63) -> element offset of its (shifted) pixel's dk-vector
    auto pixoff = [&](int t) -> size_t {
        const int y  = ((wi << 3) + (t >> 3) + SH_) & (Hd - 1);
        const int xx = ((wj << 3) + (t & 7)  + SH_) & (Wd - 1);
        return (size_t)((y << 8) + xx) * DK;
    };

    // Async-stage V: 64 tokens x 32B = 128 x 16B chunks, one per thread.
    {
        const int t  = tid >> 1;
        const int d0 = (tid & 1) * 8;
        async_copy_b128(lds_offset(&sV[t * DK + d0]),
                        qkv + vbase + pixoff(t) + d0);
    }
    async_wait0();
    __syncthreads();

    // A operand = this wave's 16 Q rows (d_k=16, zero-padded to K=32).
    v16bf aq;
    {
        const int qtok = wave * 16 + l16;
        const __bf16* qp = qkv + qbase + pixoff(qtok);
        #pragma unroll
        for (int p = 0; p < 4; ++p) {
            bf16x2 t2 = *(const bf16x2*)(qp + 2 * p + 8 * half);
            aq[2 * p] = t2.x; aq[2 * p + 1] = t2.y;
        }
        #pragma unroll
        for (int p = 4; p < 8; ++p) {
            aq[2 * p] = (__bf16)0.f; aq[2 * p + 1] = (__bf16)0.f;
        }
    }

    // Scores: 4 tiles of 16x16 = this strip's 16x64 slab.
    v8f sc[4];
    #pragma unroll
    for (int nt = 0; nt < 4; ++nt) {
        v16bf bk;
        const int ktok = nt * 16 + l16;
        const __bf16* kp = qkv + kbase + pixoff(ktok);
        #pragma unroll
        for (int p = 0; p < 4; ++p) {
            bf16x2 t2 = *(const bf16x2*)(kp + 2 * p + 8 * half);
            bk[2 * p] = t2.x; bk[2 * p + 1] = t2.y;
        }
        #pragma unroll
        for (int p = 4; p < 8; ++p) {
            bk[2 * p] = (__bf16)0.f; bk[2 * p + 1] = (__bf16)0.f;
        }
        v8f z = {};
        sc[nt] = __builtin_amdgcn_wmma_f32_16x16x32_bf16(
                     false, aq, false, bk, (short)0, z, false, false);
    }

    // Scale + relative-position bias + shift mask (f32).
    const unsigned char* mwin = mask + (size_t)(wi * 32 + wj) * 64 * 64;
    #pragma unroll
    for (int nt = 0; nt < 4; ++nt) {
        const int ktok = nt * 16 + l16;
        #pragma unroll
        for (int r = 0; r < 8; ++r) {
            const int qtok = wave * 16 + r + 8 * half;
            float s = sc[nt][r] * 0.25f +
                      rpe_table[rp_index[qtok * 64 + ktok] * NH + head];
            if (mwin[qtok * 64 + ktok]) s = -1e9f;
            sc[nt][r] = s;
        }
    }

    // Row softmax: row (r, half) lives in lanes {half*16 .. half*16+15}.
    #pragma unroll
    for (int r = 0; r < 8; ++r) {
        float m = fmaxf(fmaxf(sc[0][r], sc[1][r]), fmaxf(sc[2][r], sc[3][r]));
        #pragma unroll
        for (int off = 8; off >= 1; off >>= 1)
            m = fmaxf(m, __shfl_xor(m, off, 32));
        float s = 0.f;
        #pragma unroll
        for (int nt = 0; nt < 4; ++nt) {
            const float e = __expf(sc[nt][r] - m);
            sc[nt][r] = e; s += e;
        }
        #pragma unroll
        for (int off = 8; off >= 1; off >>= 1)
            s += __shfl_xor(s, off, 32);
        const float inv = 1.f / s;
        #pragma unroll
        for (int nt = 0; nt < 4; ++nt) sc[nt][r] *= inv;
    }

    // Stage P (bf16) to LDS (padded row-major) for A-layout regather.
    #pragma unroll
    for (int nt = 0; nt < 4; ++nt)
        #pragma unroll
        for (int r = 0; r < 8; ++r) {
            const int qtok = wave * 16 + r + 8 * half;
            sP[qtok * SP_PAD + nt * 16 + l16] = (__bf16)sc[nt][r];
        }
    __syncthreads();

    // O strip = P(16x64) @ V(64x16): two K=32 WMMA steps.
    v8f o = {};
    #pragma unroll
    for (int st = 0; st < 2; ++st) {
        v16bf ap, bv;
        #pragma unroll
        for (int p = 0; p < 8; ++p) {
            const int k = kofs(p, half) + 32 * st;
            bf16x2 t2 = *(const bf16x2*)(&sP[(wave * 16 + l16) * SP_PAD + k]);
            ap[2 * p] = t2.x; ap[2 * p + 1] = t2.y;
            bv[2 * p]     = sV[(k)     * DK + l16];   // V[k][n]
            bv[2 * p + 1] = sV[(k + 1) * DK + l16];
        }
        o = __builtin_amdgcn_wmma_f32_16x16x32_bf16(
                false, ap, false, bv, (short)0, o, false, false);
    }

    // Store with inverse cyclic shift folded into addressing.
    #pragma unroll
    for (int r = 0; r < 8; ++r) {
        const int qtok = wave * 16 + r + 8 * half;
        const int y  = ((wi << 3) + (qtok >> 3) + SH_) & (Hd - 1);
        const int xx = ((wj << 3) + (qtok & 7)  + SH_) & (Wd - 1);
        attn_out[(((size_t)b * HW) + (y << 8) + xx) * Dd + head * DK + l16] =
            (__bf16)o[r];
    }
}

// ---------------------------------------------------------------------------
// Kernel 3: output projection.  M=524288, K=96, N=96.
// One block (192 thr, 6 waves) owns a 64-row attn slab (LDS, async-staged);
// wave w -> N-tile w, reused across 4 M-strips.
// ---------------------------------------------------------------------------
#define K3_PAD 104  // bf16 per padded LDS row (208B: 16B-aligned, bank-clean)

__global__ __launch_bounds__(192) void out_proj_kernel(
    const __bf16* __restrict__ a_in, const float* __restrict__ w_out,
    const float* __restrict__ b_out, float* __restrict__ out)
{
    __shared__ __bf16 sA[64 * K3_PAD];
    const int tid  = threadIdx.x;
    const int lane = tid & 31;
    const int wave = tid >> 5;
    const int half = lane >> 4;
    const int l16  = lane & 15;
    const int row0 = blockIdx.x * 64;

    __builtin_prefetch(a_in + (size_t)(row0 + 64) * Dd + tid * 16, 0, 0);

    // Async-stage 64 rows x 96 bf16 (192B/row): 768 x 16B chunks.
    #pragma unroll
    for (int i = 0; i < 4; ++i) {
        const int c   = tid + i * 192;
        const int row = c / 12;
        const int off = c % 12;              // 16B chunk within row
        async_copy_b128(lds_offset(&sA[row * K3_PAD + off * 8]),
                        a_in + (size_t)(row0 + row) * Dd + off * 8);
    }
    async_wait0();
    __syncthreads();

    const int colB = wave * 16 + l16;        // 6 waves == 6 N-tiles

    // B operand (3 K-steps) loaded once, reused for 4 M-strips.
    v16bf bmat[3];
    #pragma unroll
    for (int kk = 0; kk < 3; ++kk)
        #pragma unroll
        for (int p = 0; p < 8; ++p) {
            const int k = kofs(p, half) + 32 * kk;
            bmat[kk][2 * p]     = (__bf16)w_out[(size_t)(k)     * Dd + colB];
            bmat[kk][2 * p + 1] = (__bf16)w_out[(size_t)(k + 1) * Dd + colB];
        }
    const float bias = b_out[colB];

    #pragma unroll
    for (int ms = 0; ms < 4; ++ms) {
        const int arow = ms * 16 + l16;
        v8f c = {};
        #pragma unroll
        for (int kk = 0; kk < 3; ++kk) {
            v16bf a;
            #pragma unroll
            for (int p = 0; p < 8; ++p) {
                const int k = kofs(p, half) + 32 * kk;
                bf16x2 t2 = *(const bf16x2*)(&sA[arow * K3_PAD + k]);
                a[2 * p] = t2.x; a[2 * p + 1] = t2.y;
            }
            c = __builtin_amdgcn_wmma_f32_16x16x32_bf16(
                    false, a, false, bmat[kk], (short)0, c, false, false);
        }
        #pragma unroll
        for (int r = 0; r < 8; ++r) {
            const int row = row0 + ms * 16 + r + 8 * half;
            out[(size_t)row * Dd + colB] = c[r] + bias;
        }
    }
}

// ---------------------------------------------------------------------------
extern "C" void kernel_launch(void* const* d_in, const int* in_sizes, int n_in,
                              void* d_out, int out_size, void* d_ws, size_t ws_size,
                              hipStream_t stream) {
    const float* x         = (const float*)d_in[0];
    const float* w_qkv     = (const float*)d_in[1];
    const float* b_qkv     = (const float*)d_in[2];
    const float* w_out     = (const float*)d_in[3];
    const float* b_out     = (const float*)d_in[4];
    const float* rpe_table = (const float*)d_in[5];
    const int*   rp_index  = (const int*)d_in[6];
    const unsigned char* mask = (const unsigned char*)d_in[7];

    // Workspace: qkv bf16 [3][B][NH][H][W][16]  then  attn bf16 [B][H][W][96]
    const size_t qkv_elems = (size_t)3 * Bd * NH * HW * DK;   // 150,994,944
    __bf16* qkv  = (__bf16*)d_ws;
    __bf16* attn = (__bf16*)((char*)d_ws + qkv_elems * sizeof(__bf16));

    // K1: 64-row slab per block
    qkv_proj_kernel<<<MROWS / 64, 192, 0, stream>>>(x, w_qkv, b_qkv, qkv);

    // K2: one block per (batch, head, window)
    win_attn_kernel<<<Bd * NH * 32 * 32, 128, 0, stream>>>(
        qkv, rpe_table, rp_index, mask, attn);

    // K3: 64-row slab per block
    out_proj_kernel<<<MROWS / 64, 192, 0, stream>>>(
        attn, w_out, b_out, (float*)d_out);
}
